// Self_Attention_matrix_nhead_76201309766438
// MI455X (gfx1250) — matrix operand order repaired and checked by reading, not compiled, not run on hardware
//
#include <hip/hip_runtime.h>

#define BATCH 4
#define SEQ   2048
#define EMBED 1024
#define NHEAD 16
#define DHEAD 64

typedef __bf16 bf16;
typedef __attribute__((ext_vector_type(16))) __bf16 v16bf;
typedef __attribute__((ext_vector_type(8)))  __bf16 v8bf;
typedef __attribute__((ext_vector_type(8)))  float  v8f;
typedef __attribute__((ext_vector_type(4)))  unsigned int u32x4;
typedef __attribute__((ext_vector_type(8)))  int i32x8;
typedef __attribute__((ext_vector_type(4)))  int i32x4;

union V16 { v16bf v; v8bf h[2]; };

// ---------------------------------------------------------------------------
// TDM helper: 2D tile load Global -> LDS, data_size = 8 bytes.
// tile_w / tile_h / stride / tensor_h are in 8-byte units.
// D# group layout per CDNA5 ISA 8.3/8.4; groups 2/3 zero (2D tensor).
// clang-23 toolchain: 6-arg builtin (extra i32x8, zero-filled).
// ---------------------------------------------------------------------------
__device__ __forceinline__ void tdm_load_2d(unsigned lds_off, const void* gaddr,
                                            unsigned tile_w, unsigned tile_h,
                                            unsigned stride, unsigned tensor_h) {
  unsigned long long ga = (unsigned long long)(uintptr_t)gaddr;
  u32x4 g0;
  g0.x = 1u;                                   // count=1, user-mode, no gather
  g0.y = lds_off;                              // lds_addr (bytes)
  g0.z = (unsigned)(ga & 0xFFFFFFFFu);         // global_addr[31:0]
  g0.w = (unsigned)((ga >> 32) & 0x01FFFFFFu)  // global_addr[56:32]
         | (2u << 30);                         // type = 2 ("image")
  unsigned td0 = stride;                       // tensor_dim0 >= tile_w
  unsigned td1 = tensor_h;                     // tensor_dim1 >= tile_h
  i32x8 g1;
  g1[0] = (int)(3u << 16);                                             // data_size=8B
  g1[1] = (int)((td0 & 0xFFFFu) << 16);                                // tdim0[15:0]
  g1[2] = (int)(((td0 >> 16) & 0xFFFFu) | ((td1 & 0xFFFFu) << 16));    // tdim0 hi | tdim1 lo
  g1[3] = (int)(((td1 >> 16) & 0xFFFFu) | ((tile_w & 0xFFFFu) << 16)); // tdim1 hi | tile_dim0
  g1[4] = (int)(tile_h & 0xFFFFu);                                     // tile_dim1 (tile_dim2=0)
  g1[5] = (int)stride;                                                 // tensor_dim0_stride lo
  g1[6] = 0;                                                           // stride hi | dim1_stride lo
  g1[7] = 0;
  i32x4 z4 = {0, 0, 0, 0};
  i32x8 z8 = {0, 0, 0, 0, 0, 0, 0, 0};
  __builtin_amdgcn_tensor_load_to_lds(g0, g1, z4, z4, z8, 0);
}

__device__ __forceinline__ unsigned lds_byte_off(const void* p) {
  // flat LDS addresses carry the LDS byte offset in the low 32 bits
  return (unsigned)(uintptr_t)p;
}

// ---------------------------------------------------------------------------
// Kernel 1: fp32 -> bf16 conversion of the input activations
// ---------------------------------------------------------------------------
__global__ void __launch_bounds__(256) convert_input_kernel(
    const float* __restrict__ x, bf16* __restrict__ out, int n) {
  int i = (blockIdx.x * blockDim.x + threadIdx.x) * 4;
  if (i + 3 < n) {
    float4 f = *(const float4*)(x + i);
    out[i + 0] = (bf16)f.x;
    out[i + 1] = (bf16)f.y;
    out[i + 2] = (bf16)f.z;
    out[i + 3] = (bf16)f.w;
  }
}

// ---------------------------------------------------------------------------
// Kernel 2: fp32 [H,E,D] weights -> bf16 transposed [H,D,E] (q,k,v packed).
// grid = (E*D/256, 3, H): no integer div/mod chains.
// ---------------------------------------------------------------------------
__global__ void __launch_bounds__(256) convert_w_kernel(
    const float* __restrict__ wq, const float* __restrict__ wk,
    const float* __restrict__ wv, bf16* __restrict__ wt) {
  const int which = blockIdx.y;
  const int h     = blockIdx.z;
  int idx = blockIdx.x * blockDim.x + threadIdx.x;   // over E*D
  int e = idx >> 6;          // / DHEAD
  int d = idx & 63;          // % DHEAD
  const float* src = (which == 0) ? wq : (which == 1) ? wk : wv;
  float v = src[(size_t)h * EMBED * DHEAD + (size_t)e * DHEAD + d];
  const size_t per = (size_t)NHEAD * EMBED * DHEAD;
  wt[(size_t)which * per + ((size_t)h * DHEAD + d) * EMBED + e] = (bf16)v;
}

// ---------------------------------------------------------------------------
// Kernel 3 helpers: load one K-step fragment set / issue its WMMA chain.
// ---------------------------------------------------------------------------
__device__ __forceinline__ void load_set(
    const bf16* arow, const bf16* w0, const bf16* w1,
    const bf16* w2, const bf16* w3, int ko, int kA,
    V16& a, V16& b0, V16& b1, V16& b2, V16& b3) {
  a.h[0]  = *(const v8bf*)(arow + ko + kA);
  a.h[1]  = *(const v8bf*)(arow + ko + kA + 8);
  b0.h[0] = *(const v8bf*)(w0 + ko);
  b0.h[1] = *(const v8bf*)(w0 + ko + 8);
  b1.h[0] = *(const v8bf*)(w1 + ko);
  b1.h[1] = *(const v8bf*)(w1 + ko + 8);
  b2.h[0] = *(const v8bf*)(w2 + ko);
  b2.h[1] = *(const v8bf*)(w2 + ko + 8);
  b3.h[0] = *(const v8bf*)(w3 + ko);
  b3.h[1] = *(const v8bf*)(w3 + ko + 8);
}

__device__ __forceinline__ void wmma_set(
    const V16& a, const V16& b0, const V16& b1, const V16& b2, const V16& b3,
    v8f& acc0, v8f& acc1, v8f& acc2, v8f& acc3) {
  acc0 = __builtin_amdgcn_wmma_f32_16x16x32_bf16(false, a.v, false, b0.v,
                                                 (short)0, acc0, false, false);
  acc1 = __builtin_amdgcn_wmma_f32_16x16x32_bf16(false, a.v, false, b1.v,
                                                 (short)0, acc1, false, false);
  acc2 = __builtin_amdgcn_wmma_f32_16x16x32_bf16(false, a.v, false, b2.v,
                                                 (short)0, acc2, false, false);
  acc3 = __builtin_amdgcn_wmma_f32_16x16x32_bf16(false, a.v, false, b3.v,
                                                 (short)0, acc3, false, false);
}

// ---------------------------------------------------------------------------
// Kernel 3: QKV projection GEMM [S x E] @ [E x D].  One wave -> 16 x 64 tile.
// Software-pipelined: K-loop unrolled x2 with ping/pong register sets so
// loads for step i+1 are in flight during the WMMA chain of step i.
// ---------------------------------------------------------------------------
__global__ void __launch_bounds__(128) proj_kernel(
    const bf16* __restrict__ Xb, const bf16* __restrict__ Wt,
    bf16* __restrict__ Q, bf16* __restrict__ K, bf16* __restrict__ Vt) {
  const int lane = threadIdx.x & 31;
  const int wave = threadIdx.x >> 5;

  int idx = blockIdx.x;
  const int MB = SEQ / 64;
  int mblk  = idx % MB;  idx /= MB;
  int which = idx % 3;   idx /= 3;
  int h     = idx % NHEAD;
  int b     = idx / NHEAD;

  const int mbase = mblk * 64 + wave * 16;
  const bf16* Xp = Xb + (size_t)b * SEQ * EMBED;
  const bf16* Wp = Wt + ((size_t)which * NHEAD + h) * (size_t)DHEAD * EMBED;

  const int kA  = (lane < 16) ? 0 : 16;
  const int kB  = (lane < 16) ? 0 : 16;
  const int n15 = lane & 15;

  const bf16* arow  = Xp + (size_t)(mbase + (lane & 15)) * EMBED;
  const bf16* wrow0 = Wp + (size_t)(0 * 16 + n15) * EMBED + kB;
  const bf16* wrow1 = Wp + (size_t)(1 * 16 + n15) * EMBED + kB;
  const bf16* wrow2 = Wp + (size_t)(2 * 16 + n15) * EMBED + kB;
  const bf16* wrow3 = Wp + (size_t)(3 * 16 + n15) * EMBED + kB;

  v8f acc0 = {}, acc1 = {}, acc2 = {}, acc3 = {};
  V16 aA, bA0, bA1, bA2, bA3;
  V16 aB, bB0, bB1, bB2, bB3;

  load_set(arow, wrow0, wrow1, wrow2, wrow3, 0, kA, aA, bA0, bA1, bA2, bA3);
  for (int ko = 0; ko < EMBED; ko += 64) {
    load_set(arow, wrow0, wrow1, wrow2, wrow3, ko + 32, kA,
             aB, bB0, bB1, bB2, bB3);
    wmma_set(aA, bA0, bA1, bA2, bA3, acc0, acc1, acc2, acc3);
    if (ko + 64 < EMBED)
      load_set(arow, wrow0, wrow1, wrow2, wrow3, ko + 64, kA,
               aA, bA0, bA1, bA2, bA3);
    wmma_set(aB, bB0, bB1, bB2, bB3, acc0, acc1, acc2, acc3);
  }

  // destination select once (uniform), then unconditional stores
  const size_t bh = (size_t)(b * NHEAD + h);
  bf16* dst;
  size_t rs, cs;
  if (which == 0)      { dst = Q  + bh * (size_t)SEQ * DHEAD; rs = DHEAD; cs = 1; }
  else if (which == 1) { dst = K  + bh * (size_t)SEQ * DHEAD; rs = DHEAD; cs = 1; }
  else                 { dst = Vt + bh * (size_t)DHEAD * SEQ; rs = 1;     cs = SEQ; }

  const int rowoff = (lane < 16) ? 0 : 8;
  v8f accs[4] = {acc0, acc1, acc2, acc3};
#pragma unroll
  for (int j = 0; j < 4; ++j) {
#pragma unroll
    for (int r = 0; r < 8; ++r) {
      size_t row = (size_t)(mbase + r + rowoff);
      size_t col = (size_t)(j * 16 + n15);
      dst[row * rs + col * cs] = (bf16)accs[j][r];
    }
  }
}

// ---------------------------------------------------------------------------
// Kernel 4: flash attention with TDM-staged, double-buffered K/V tiles.
// Block = 4 waves sharing (b,h) and a 32-key window; wave0 drives the TDM:
//   K tile: 32 rows x 128B contiguous   -> kbuf (2D: 16 x 32 in 8B units)
//   V tile: 64 rows x 64B, stride 4096B -> vbuf (2D: 8 x 64, stride 512)
// s_wait_tensorcnt(2) overlaps the next tile's DMA with current compute.
// ---------------------------------------------------------------------------
__global__ void __launch_bounds__(128) flash_kernel(
    const bf16* __restrict__ Q, const bf16* __restrict__ K,
    const bf16* __restrict__ Vt, float* __restrict__ out) {
  __shared__ __align__(256) bf16 kbuf[2][32 * 64];
  __shared__ __align__(256) bf16 vbuf[2][64 * 32];
  __shared__ __align__(16)  bf16 pbuf[4][16 * 32];

  const int lane = threadIdx.x & 31;
  const int wave = threadIdx.x >> 5;

  int idx = blockIdx.x;
  const int QB = SEQ / 64;
  int qblk = idx % QB; idx /= QB;
  int h    = idx % NHEAD;
  int b    = idx / NHEAD;

  const int qbase = qblk * 64 + wave * 16;
  const size_t bh = (size_t)(b * NHEAD + h);
  const bf16* Qp = Q  + bh * (size_t)SEQ * DHEAD;
  const bf16* Kp = K  + bh * (size_t)SEQ * DHEAD;
  const bf16* Vp = Vt + bh * (size_t)DHEAD * SEQ;

  const int m      = lane & 15;
  const int kA     = (lane < 16) ? 0 : 16;
  const int kB     = (lane < 16) ? 0 : 16;
  const int n15    = lane & 15;
  const int rowoff = (lane < 16) ? 0 : 8;

  // Q tile in A-layout (global, once per wave)
  const bf16* qrow = Qp + (size_t)(qbase + m) * DHEAD;
  V16 a0, a1;
  a0.h[0] = *(const v8bf*)(qrow + kA);
  a0.h[1] = *(const v8bf*)(qrow + kA + 8);
  a1.h[0] = *(const v8bf*)(qrow + 32 + kA);
  a1.h[1] = *(const v8bf*)(qrow + 32 + kA + 8);

  v8f o[4] = {{}, {}, {}, {}};
  float rmax[8], rsum[8];
#pragma unroll
  for (int r = 0; r < 8; ++r) { rmax[r] = -3.0e38f; rsum[r] = 0.0f; }

  // prologue: DMA first K/V tile into buffer 0
  if (wave == 0) {
    tdm_load_2d(lds_byte_off(&kbuf[0][0]), Kp, 16, 32, 16, 32);
    tdm_load_2d(lds_byte_off(&vbuf[0][0]), Vp, 8, 64, SEQ * 2 / 8, 64);
  }

  int cur = 0;
  for (int kb = 0; kb < SEQ; kb += 32) {
    if (wave == 0) {
      if (kb + 32 < SEQ) {
        const int nxt = cur ^ 1;
        tdm_load_2d(lds_byte_off(&kbuf[nxt][0]),
                    Kp + (size_t)(kb + 32) * DHEAD, 16, 32, 16, 32);
        tdm_load_2d(lds_byte_off(&vbuf[nxt][0]),
                    Vp + (kb + 32), 8, 64, SEQ * 2 / 8, 64);
        __builtin_amdgcn_s_wait_tensorcnt(2);   // older buffer's 2 ops done
      } else {
        __builtin_amdgcn_s_wait_tensorcnt(0);
      }
    }
    __syncthreads();   // publish buffer `cur` to all waves

    const bf16* kl = &kbuf[cur][0];
    const bf16* vl = &vbuf[cur][0];

    // ---- scores: two 16-key tiles, inner dim d=64 (2 x K=32 each) ----
    v8f s0 = {}, s1 = {};
    {
      const bf16* kr = kl + (size_t)n15 * 64 + kB;
      V16 blo, bhi;
      blo.h[0] = *(const v8bf*)(kr);       blo.h[1] = *(const v8bf*)(kr + 8);
      bhi.h[0] = *(const v8bf*)(kr + 32);  bhi.h[1] = *(const v8bf*)(kr + 40);
      s0 = __builtin_amdgcn_wmma_f32_16x16x32_bf16(false, a0.v, false, blo.v,
                                                   (short)0, s0, false, false);
      s0 = __builtin_amdgcn_wmma_f32_16x16x32_bf16(false, a1.v, false, bhi.v,
                                                   (short)0, s0, false, false);
    }
    {
      const bf16* kr = kl + (size_t)(16 + n15) * 64 + kB;
      V16 blo, bhi;
      blo.h[0] = *(const v8bf*)(kr);       blo.h[1] = *(const v8bf*)(kr + 8);
      bhi.h[0] = *(const v8bf*)(kr + 32);  bhi.h[1] = *(const v8bf*)(kr + 40);
      s1 = __builtin_amdgcn_wmma_f32_16x16x32_bf16(false, a0.v, false, blo.v,
                                                   (short)0, s1, false, false);
      s1 = __builtin_amdgcn_wmma_f32_16x16x32_bf16(false, a1.v, false, bhi.v,
                                                   (short)0, s1, false, false);
    }

    // ---- online softmax: per-row max/sum via 16-lane half reductions ----
    float alpha[8];
#pragma unroll
    for (int r = 0; r < 8; ++r) {
      float mx = fmaxf(s0[r], s1[r]);
      mx = fmaxf(mx, __shfl_xor(mx, 1, 32));
      mx = fmaxf(mx, __shfl_xor(mx, 2, 32));
      mx = fmaxf(mx, __shfl_xor(mx, 4, 32));
      mx = fmaxf(mx, __shfl_xor(mx, 8, 32));
      float nm = fmaxf(rmax[r], mx);
      alpha[r] = __expf(rmax[r] - nm);
      rmax[r]  = nm;
      float e0 = __expf(s0[r] - nm);
      float e1 = __expf(s1[r] - nm);
      s0[r] = e0;
      s1[r] = e1;
      float ts = e0 + e1;
      ts += __shfl_xor(ts, 1, 32);
      ts += __shfl_xor(ts, 2, 32);
      ts += __shfl_xor(ts, 4, 32);
      ts += __shfl_xor(ts, 8, 32);
      rsum[r] = rsum[r] * alpha[r] + ts;
    }
#pragma unroll
    for (int j = 0; j < 4; ++j)
#pragma unroll
      for (int r = 0; r < 8; ++r) o[j][r] *= alpha[r];

    // ---- P (C-layout f32) -> bf16 in LDS, reload in A-layout ----
#pragma unroll
    for (int r = 0; r < 8; ++r) {
      int row = r + rowoff;
      pbuf[wave][row * 32 + n15]      = (bf16)s0[r];
      pbuf[wave][row * 32 + 16 + n15] = (bf16)s1[r];
    }
    V16 ap;
    ap.h[0] = *(const v8bf*)&pbuf[wave][m * 32 + kA];
    ap.h[1] = *(const v8bf*)&pbuf[wave][m * 32 + kA + 8];

    // ---- O += P @ V (V tile in LDS, rows = d, contiguous along s) ----
#pragma unroll
    for (int j = 0; j < 4; ++j) {
      const bf16* vr = vl + (size_t)(j * 16 + n15) * 32 + kB;
      V16 bv;
      bv.h[0] = *(const v8bf*)(vr);
      bv.h[1] = *(const v8bf*)(vr + 8);
      o[j] = __builtin_amdgcn_wmma_f32_16x16x32_bf16(false, ap.v, false, bv.v,
                                                     (short)0, o[j], false, false);
    }

    __syncthreads();   // all waves done with buffer `cur` before it is reused
    cur ^= 1;
  }

  // ---- epilogue: divide by rowsum * sqrt(DHEAD), write [B,S,H*D] fp32 ----
  float inv[8];
#pragma unroll
  for (int r = 0; r < 8; ++r) inv[r] = 1.0f / (rsum[r] * 8.0f);
#pragma unroll
  for (int j = 0; j < 4; ++j)
#pragma unroll
    for (int r = 0; r < 8; ++r) {
      int row = qbase + r + rowoff;
      int col = h * DHEAD + j * 16 + n15;
      out[((size_t)b * SEQ + row) * EMBED + col] = o[j][r] * inv[r];
    }
}

// ---------------------------------------------------------------------------
extern "C" void kernel_launch(void* const* d_in, const int* in_sizes, int n_in,
                              void* d_out, int out_size, void* d_ws, size_t ws_size,
                              hipStream_t stream) {
  const float* x  = (const float*)d_in[0];
  const float* wq = (const float*)d_in[1];
  const float* wk = (const float*)d_in[2];
  const float* wv = (const float*)d_in[3];
  float* out = (float*)d_out;

  const size_t nX = (size_t)BATCH * SEQ * EMBED;
  const size_t nW = (size_t)NHEAD * EMBED * DHEAD;
  const size_t nQ = (size_t)BATCH * NHEAD * SEQ * DHEAD;

  bf16* Xb  = (bf16*)d_ws;
  bf16* Wt  = Xb + nX;
  bf16* Qb  = Wt + 3 * nW;
  bf16* Kb  = Qb + nQ;
  bf16* Vtb = Kb + nQ;

  convert_input_kernel<<<(int)(nX / 1024), 256, 0, stream>>>(x, Xb, (int)nX);
  dim3 wgrid((EMBED * DHEAD) / 256, 3, NHEAD);
  convert_w_kernel<<<wgrid, 256, 0, stream>>>(wq, wk, wv, Wt);
  proj_kernel<<<BATCH * NHEAD * 3 * (SEQ / 64), 128, 0, stream>>>(Xb, Wt, Qb, Kb, Vtb);
  flash_kernel<<<BATCH * NHEAD * (SEQ / 64), 128, 0, stream>>>(Qb, Kb, Vtb, out);
}
